// ImplicitDecoder_2276332667648
// MI455X (gfx1250) — matrix operand order repaired and checked by reading, not compiled, not run on hardware
//
#include <hip/hip_runtime.h>
#include <hip/hip_bf16.h>
#include <math.h>

typedef __hip_bfloat16 bf16;
typedef __attribute__((ext_vector_type(16))) __bf16        v16bf;
typedef __attribute__((ext_vector_type(8)))  float         v8f;
typedef __attribute__((ext_vector_type(8)))  unsigned int  v8u;

#define M_PIX 65536   // 256*256 output pixels
#define HIN   96
#define WIN   96

// Epilogue modes (compile-time)
#define EPI_NONE   0
#define EPI_RELU   1   // bf16 out = relu(acc+b)
#define EPI_SINMUL 2   // bf16 out = aux * sin(acc+b)
#define EPI_ADDAUX 3   // bf16 out = acc+b + aux      (residual)
#define EPI_BRF32  4   // f32  OutF[p*ldo+n] = relu(acc+b)   (branch tail)
#define EPI_FINAL  5   // f32  OutF[n*M+p]   = acc+b + AuxF[p*ldaux+n]

// ---- CDNA5 async global->LDS DMA (ASYNCcnt path), with sync fallback ------
#if defined(__has_builtin)
#  if __has_builtin(__builtin_amdgcn_global_load_async_to_lds_b128)
#    define USE_ASYNC_LDS 1
#  endif
#endif

#ifdef USE_ASYNC_LDS
typedef int v4i __attribute__((vector_size(16)));
typedef __attribute__((address_space(1))) v4i* gv4;   // global int4*
typedef __attribute__((address_space(3))) v4i* lv4;   // LDS int4*
#endif

__device__ __forceinline__ void copy16_g2l(const bf16* g, bf16* l) {
#ifdef USE_ASYNC_LDS
  __builtin_amdgcn_global_load_async_to_lds_b128((gv4)g, (lv4)l, 0, 0);
#else
  *(uint4*)l = *(const uint4*)g;
  __builtin_prefetch(g + 32, 0, 1);          // speculative global_prefetch
#endif
}

__device__ __forceinline__ void wait_async_all() {
#ifdef USE_ASYNC_LDS
#  if __has_builtin(__builtin_amdgcn_s_wait_asynccnt)
  __builtin_amdgcn_s_wait_asynccnt(0);
#  else
  asm volatile("s_wait_asynccnt 0x0" ::: "memory");
#  endif
#endif
}

// ---------------------------------------------------------------------------
// f32 -> bf16 weight conversion with zero padding to (Rp x Cp)
// ---------------------------------------------------------------------------
__global__ void k_cvt_pad(const float* __restrict__ s, bf16* __restrict__ d,
                          int R, int C, int Rp, int Cp) {
  int i = blockIdx.x * 256 + threadIdx.x;
  if (i >= Rp * Cp) return;
  int r = i / Cp, c = i - r * Cp;
  float v = (r < R && c < C) ? s[r * C + c] : 0.f;
  d[i] = __float2bfloat16(v);
}

// ---------------------------------------------------------------------------
// Fused unfold3 (3x3, pad 1) + bilinear upsample 96x96 -> 256x256.
// Output layout: U[pixel][c*9 + tap]  (bf16, K-contiguous for GEMM A tiles)
// ---------------------------------------------------------------------------
__global__ void k_unfold_bilinear(const float* __restrict__ X0, const float* __restrict__ X1,
                                  bf16* __restrict__ U0, bf16* __restrict__ U1) {
  const float* X = blockIdx.y ? X1 : X0;
  bf16*        U = blockIdx.y ? U1 : U0;
  int idx = blockIdx.x * blockDim.x + threadIdx.x;
  if (idx >= M_PIX * 64) return;
  int c = idx & 63, p = idx >> 6;
  int ho = p >> 8, wo = p & 255;
  float ch = fminf(fmaxf((ho + 0.5f) * (96.f / 256.f) - 0.5f, 0.f), 95.f);
  float cw = fminf(fmaxf((wo + 0.5f) * (96.f / 256.f) - 0.5f, 0.f), 95.f);
  int   h0 = (int)ch, w0 = (int)cw;
  float lh = ch - (float)h0, lw = cw - (float)w0;
  const float* Xc = X + (size_t)c * (HIN * WIN);
  float g[4][4];
#pragma unroll
  for (int r = 0; r < 4; ++r) {
    int h = h0 - 1 + r;
#pragma unroll
    for (int s = 0; s < 4; ++s) {
      int w = w0 - 1 + s;
      g[r][s] = (h >= 0 && h < HIN && w >= 0 && w < WIN) ? Xc[h * WIN + w] : 0.f;
    }
  }
  bf16* out = U + (size_t)p * 576 + c * 9;
#pragma unroll
  for (int i = 0; i < 3; ++i)
#pragma unroll
    for (int j = 0; j < 3; ++j) {
      float v = (1.f - lh) * ((1.f - lw) * g[i][j]     + lw * g[i][j + 1])
              +        lh  * ((1.f - lw) * g[i + 1][j] + lw * g[i + 1][j + 1]);
      out[i * 3 + j] = __float2bfloat16(v);
    }
}

// ---------------------------------------------------------------------------
// Positional encodings, K-padded to 32: S[pixel][32] = {rel_h, rel_w, r, r, 0...}
// ---------------------------------------------------------------------------
__global__ void k_syn(bf16* __restrict__ S) {
  int p = blockIdx.x * blockDim.x + threadIdx.x;
  if (p >= M_PIX) return;
  int ho = p >> 8, wo = p & 255;
  int ih = min((int)((ho + 0.5f) * 0.375f), 95);
  int iw = min((int)((wo + 0.5f) * 0.375f), 95);
  float hup = -1.f + 1.f / 256.f + (2.f / 256.f) * (float)ho;
  float wup = -1.f + 1.f / 256.f + (2.f / 256.f) * (float)wo;
  float hin = -1.f + 1.f / 96.f + (2.f / 96.f) * (float)ih;
  float win = -1.f + 1.f / 96.f + (2.f / 96.f) * (float)iw;
  bf16* o = S + (size_t)p * 32;
  o[0] = __float2bfloat16((hup - hin) * 96.f);
  o[1] = __float2bfloat16((wup - win) * 96.f);
  o[2] = __float2bfloat16(0.375f);
  o[3] = __float2bfloat16(0.375f);
#pragma unroll
  for (int i = 4; i < 32; ++i) o[i] = __float2bfloat16(0.f);
}

// ---------------------------------------------------------------------------
// 1x1-conv GEMM, compile-time shapes + epilogue.
//   Out[p,n] = EPI( sum_k A[p,k]*W[n,k] + bias[n] )
//   A: [M_PIX, K] bf16 (concat of A|A2 split at KSPLIT when KSPLIT != 0)
//   W: [N, K] bf16 (torch layout, acts as B^T); K % 32 == 0, N % 16 == 0.
// Block: 256 thr = 8 waves; wave = 16-pixel M strip x N columns (N/16 tiles).
// Double-buffered LDS staging via async global->LDS DMA overlapped with WMMA;
// B fragments software-pipelined one stage ahead of the WMMA chain.
// ---------------------------------------------------------------------------
template <int K, int N, int KSPLIT, int EPI>
__global__ __launch_bounds__(256)
void k_gemm(const bf16* __restrict__ A, const bf16* __restrict__ A2,
            const bf16* __restrict__ W, const float* __restrict__ bias,
            int nOut,
            bf16* __restrict__ Out, int ldo, int ooff,
            const bf16* __restrict__ aux, int ldaux, int aoff,
            float* __restrict__ OutF, const float* __restrict__ AuxF)
{
  constexpr int NT  = N / 16;                 // WMMA tiles along N
  constexpr int LDA = KSPLIT ? KSPLIT : K;    // row stride of A (and A2)

  __shared__ bf16 lA[2][128 * 32];            // [buf][row][k], 64B rows
  __shared__ bf16 lB[2][N * 32];              // [buf][n][k]

  const int tid   = threadIdx.x;
  const int lane  = tid & 31;
  const int wave  = tid >> 5;
  const int m_l   = lane & 15;
  const int hi    = lane >> 4;
  const int pblk  = blockIdx.x * 128;         // block M base
  const int pbase = pblk + wave * 16;         // wave M base

  // stage one 32-K slice into LDS buffer `buf`
  auto stage = [&](int kb, int buf) {
    const bf16* srcA = A;
    int kbA = kb;
    if (KSPLIT && kb >= KSPLIT) { srcA = A2; kbA = kb - KSPLIT; }  // uniform
#pragma unroll
    for (int i = 0; i < 2; ++i) {             // A: 128 rows x 64B = 512 chunks
      int c   = tid + i * 256;
      int row = c >> 2;
      int off = (c & 3) * 8;                  // bf16 elements
      copy16_g2l(srcA + (size_t)(pblk + row) * LDA + kbA + off,
                 &lA[buf][row * 32 + off]);
    }
    for (int c = tid; c < N * 4; c += 256) {  // B: N rows x 64B
      int n   = c >> 2;
      int off = (c & 3) * 8;
      copy16_g2l(W + (size_t)n * K + kb + off, &lB[buf][n * 32 + off]);
    }
  };

  v8f acc[NT];
#pragma unroll
  for (int t = 0; t < NT; ++t) acc[t] = (v8f){0.f, 0.f, 0.f, 0.f, 0.f, 0.f, 0.f, 0.f};

  stage(0, 0);
  wait_async_all();
  __syncthreads();

  int buf = 0;
  for (int kb = 0; kb < K; kb += 32) {
    // kick off DMA for the next slice into the other buffer
    if (kb + 32 < K) stage(kb + 32, buf ^ 1);

    // ---- A fragment (16x32): v0..3 = k hi*8+0..7, v4..7 = k 16+hi*8+0..7 --
    const int r = wave * 16 + m_l;
    uint4 alo = *(const uint4*)&lA[buf][r * 32 + hi * 8];
    uint4 ahi = *(const uint4*)&lA[buf][r * 32 + 16 + hi * 8];
    v8u au; au[0]=alo.x; au[1]=alo.y; au[2]=alo.z; au[3]=alo.w;
            au[4]=ahi.x; au[5]=ahi.y; au[6]=ahi.z; au[7]=ahi.w;
    v16bf afrag = __builtin_bit_cast(v16bf, au);

    // ---- B fragments (32x16), pipelined one stage ahead of the WMMAs ----
    auto ldb = [&](int t) -> v8u {
      const int n = t * 16 + m_l;
      uint4 blo = *(const uint4*)&lB[buf][n * 32 + hi * 8];
      uint4 bhi = *(const uint4*)&lB[buf][n * 32 + 16 + hi * 8];
      v8u b; b[0]=blo.x; b[1]=blo.y; b[2]=blo.z; b[3]=blo.w;
             b[4]=bhi.x; b[5]=bhi.y; b[6]=bhi.z; b[7]=bhi.w;
      return b;
    };
    v8u bcur = ldb(0);
#pragma unroll
    for (int t = 0; t < NT; ++t) {
      v8u bnxt = (t + 1 < NT) ? ldb(t + 1) : bcur;   // issue loads before WMMA t
      acc[t] = __builtin_amdgcn_wmma_f32_16x16x32_bf16(
          false, afrag, false, __builtin_bit_cast(v16bf, bcur),
          (short)0, acc[t], false, false);
      bcur = bnxt;
    }

    wait_async_all();     // next buffer's DMA complete (this wave)
    __syncthreads();      // all waves done reading buf / filling buf^1
    buf ^= 1;
  }

  // ---- epilogue: C/D layout n = lane&15, row m = v + 8*hi ----
#pragma unroll
  for (int t = 0; t < NT; ++t) {
    int n = t * 16 + m_l;
    if (n < nOut) {
      float bv = bias[n];
#pragma unroll
      for (int v = 0; v < 8; ++v) {
        int   p   = pbase + hi * 8 + v;
        float val = acc[t][v] + bv;
        if (EPI == EPI_RELU) {
          Out[(size_t)p * ldo + ooff + n] = __float2bfloat16(fmaxf(val, 0.f));
        } else if (EPI == EPI_SINMUL) {
          float m = __bfloat162float(aux[(size_t)p * ldaux + aoff + n]);
          Out[(size_t)p * ldo + ooff + n] = __float2bfloat16(m * __sinf(val));
        } else if (EPI == EPI_ADDAUX) {
          val += __bfloat162float(aux[(size_t)p * ldaux + aoff + n]);
          Out[(size_t)p * ldo + ooff + n] = __float2bfloat16(val);
        } else if (EPI == EPI_BRF32) {
          OutF[(size_t)p * ldo + n] = fmaxf(val, 0.f);
        } else if (EPI == EPI_FINAL) {
          OutF[(size_t)n * M_PIX + p] = val + AuxF[(size_t)p * ldaux + n];
        } else {
          Out[(size_t)p * ldo + ooff + n] = __float2bfloat16(val);
        }
      }
    }
  }
}

template <int K, int N, int KSPLIT, int EPI>
static void launch_gemm(hipStream_t s,
                        const bf16* A, const bf16* A2, const bf16* W,
                        const float* bias, int nOut,
                        bf16* Out, int ldo, int ooff,
                        const bf16* aux, int ldaux, int aoff,
                        float* OutF, const float* AuxF) {
  k_gemm<K, N, KSPLIT, EPI><<<dim3(M_PIX / 128), 256, 0, s>>>(
      A, A2, W, bias, nOut, Out, ldo, ooff, aux, ldaux, aoff, OutF, AuxF);
}

// ---------------------------------------------------------------------------
// Host orchestration
// ---------------------------------------------------------------------------
extern "C" void kernel_launch(void* const* d_in, const int* in_sizes, int n_in,
                              void* d_out, int out_size, void* d_ws, size_t ws_size,
                              hipStream_t stream) {
  (void)in_sizes; (void)n_in; (void)out_size; (void)ws_size;
  const int M = M_PIX;

  const float* x   = (const float*)d_in[0];
  const float* ref = (const float*)d_in[1];

  // params flattened in reference insertion order:
  // K[5]{in.w,in.b,(res.w,res.b)x4}, Q[5]{w,b}, last{w,b}, in_branch x3{w,b}, ref_branch x3{w,b}
  const float *KinWf[5], *KinB[5], *ResWf[5][4], *ResB[5][4];
  for (int L = 0; L < 5; ++L) {
    int b = 2 + 10 * L;
    KinWf[L] = (const float*)d_in[b];
    KinB[L]  = (const float*)d_in[b + 1];
    for (int r = 0; r < 4; ++r) {
      ResWf[L][r] = (const float*)d_in[b + 2 + 2 * r];
      ResB[L][r]  = (const float*)d_in[b + 3 + 2 * r];
    }
  }
  const float *QWf[5], *QB[5];
  for (int L = 0; L < 5; ++L) { QWf[L] = (const float*)d_in[52 + 2 * L]; QB[L] = (const float*)d_in[53 + 2 * L]; }
  const float* LastWf = (const float*)d_in[62];
  const float* LastB  = (const float*)d_in[63];
  const float *IbWf[3], *IbB[3], *RbWf[3], *RbB[3];
  for (int i = 0; i < 3; ++i) {
    IbWf[i] = (const float*)d_in[64 + 2 * i]; IbB[i] = (const float*)d_in[65 + 2 * i];
    RbWf[i] = (const float*)d_in[70 + 2 * i]; RbB[i] = (const float*)d_in[71 + 2 * i];
  }

  // ---- workspace carve ----
  char*  ws  = (char*)d_ws;
  size_t off = 0;
  auto alloc = [&](size_t bytes) -> char* {
    char* p = ws + off;
    off = (off + bytes + 255) & ~(size_t)255;
    return p;
  };
  // convert f32 weight (R x C) into zero-padded bf16 (Rp x Cp)
  auto cvt = [&](const float* src, int R, int C, int Rp, int Cp) -> bf16* {
    bf16* d = (bf16*)alloc((size_t)Rp * Cp * sizeof(bf16));
    int n = Rp * Cp;
    k_cvt_pad<<<(n + 255) / 256, 256, 0, stream>>>(src, d, R, C, Rp, Cp);
    return d;
  };

  // bf16 weights (padded where the GEMM shape requires it)
  bf16 *KinW[5], *ResW[5][4], *QW[5], *LastW, *IbW[3], *RbW[3];
  KinW[0] = cvt(KinWf[0], 128, 1152, 128, 1152);
  for (int L = 1; L < 5; ++L) KinW[L] = cvt(KinWf[L], 128, 128, 128, 128);
  for (int L = 0; L < 5; ++L)
    for (int r = 0; r < 4; ++r) ResW[L][r] = cvt(ResWf[L][r], 128, 128, 128, 128);
  QW[0] = cvt(QWf[0], 64, 4, 64, 32);               // K pad 4 -> 32
  for (int L = 1; L < 5; ++L) QW[L] = cvt(QWf[L], 64, 64, 64, 64);
  LastW = cvt(LastWf, 2, 64, 16, 64);               // N pad 2 -> 16
  IbW[0] = cvt(IbWf[0], 64, 576, 64, 576);
  IbW[1] = cvt(IbWf[1], 64, 64, 64, 64);
  IbW[2] = cvt(IbWf[2], 2, 64, 16, 64);             // N pad 2 -> 16
  RbW[0] = cvt(RbWf[0], 64, 576, 64, 576);
  RbW[1] = cvt(RbWf[1], 64, 64, 64, 64);
  RbW[2] = cvt(RbWf[2], 2, 64, 16, 64);             // N pad 2 -> 16

  // activation buffers (bf16 unless noted)
  bf16* XU    = (bf16*)alloc((size_t)M * 576 * 2);
  bf16* RU    = (bf16*)alloc((size_t)M * 576 * 2);
  bf16* SYN   = (bf16*)alloc((size_t)M * 32 * 2);   // K-padded pos-enc
  bf16* KKa   = (bf16*)alloc((size_t)M * 128 * 2);
  bf16* KKb   = (bf16*)alloc((size_t)M * 128 * 2);
  bf16* T128a = (bf16*)alloc((size_t)M * 128 * 2);
  bf16* T128b = (bf16*)alloc((size_t)M * 128 * 2);
  bf16* qbuf[2] = {(bf16*)alloc((size_t)M * 64 * 2), (bf16*)alloc((size_t)M * 64 * 2)};
  bf16* rbuf[2] = {(bf16*)alloc((size_t)M * 64 * 2), (bf16*)alloc((size_t)M * 64 * 2)};
  bf16* T64a = (bf16*)alloc((size_t)M * 64 * 2);
  bf16* T64b = (bf16*)alloc((size_t)M * 64 * 2);
  float* BRx = (float*)alloc((size_t)M * 2 * 4);
  float* BRr = (float*)alloc((size_t)M * 2 * 4);

  // ---- prep: fused unfold+bilinear for x and ref; positional encodings ----
  dim3 gp((M * 64 + 255) / 256, 2);
  k_unfold_bilinear<<<gp, 256, 0, stream>>>(x, ref, XU, RU);
  k_syn<<<(M + 255) / 256, 256, 0, stream>>>(SYN);

  const bf16* NB = nullptr;
  float*      NF = nullptr;
  const float* NCF = nullptr;

  // ---- main K/Q tower ----
  for (int L = 0; L < 5; ++L) {
    // kk = relu(conv_in(kk))
    if (L == 0)
      launch_gemm<1152, 128, 576, EPI_RELU>(stream, XU, RU, KinW[0], KinB[0], 128,
                                            KKa, 128, 0, NB, 0, 0, NF, NCF);
    else
      launch_gemm<128, 128, 0, EPI_RELU>(stream, KKb, NB, KinW[L], KinB[L], 128,
                                         KKa, 128, 0, NB, 0, 0, NF, NCF);
    // resblock: 3x (conv+relu), conv, + residual(KKa)
    launch_gemm<128, 128, 0, EPI_RELU>(stream, KKa,   NB, ResW[L][0], ResB[L][0], 128,
                                       T128a, 128, 0, NB, 0, 0, NF, NCF);
    launch_gemm<128, 128, 0, EPI_RELU>(stream, T128a, NB, ResW[L][1], ResB[L][1], 128,
                                       T128b, 128, 0, NB, 0, 0, NF, NCF);
    launch_gemm<128, 128, 0, EPI_RELU>(stream, T128b, NB, ResW[L][2], ResB[L][2], 128,
                                       T128a, 128, 0, NB, 0, 0, NF, NCF);
    launch_gemm<128, 128, 0, EPI_ADDAUX>(stream, T128a, NB, ResW[L][3], ResB[L][3], 128,
                                         KKb, 128, 0, KKa, 128, 0, NF, NCF);
    // q = kk[:, :64] * sin(conv(q)); q_ref = kk[:, 64:] * sin(conv(q_ref))
    if (L == 0) {
      launch_gemm<32, 64, 0, EPI_SINMUL>(stream, SYN, NB, QW[0], QB[0], 64,
                                         qbuf[0], 64, 0, KKb, 128, 0, NF, NCF);
      launch_gemm<32, 64, 0, EPI_SINMUL>(stream, SYN, NB, QW[0], QB[0], 64,
                                         rbuf[0], 64, 0, KKb, 128, 64, NF, NCF);
    } else {
      launch_gemm<64, 64, 0, EPI_SINMUL>(stream, qbuf[(L + 1) & 1], NB, QW[L], QB[L], 64,
                                         qbuf[L & 1], 64, 0, KKb, 128, 0, NF, NCF);
      launch_gemm<64, 64, 0, EPI_SINMUL>(stream, rbuf[(L + 1) & 1], NB, QW[L], QB[L], 64,
                                         rbuf[L & 1], 64, 0, KKb, 128, 64, NF, NCF);
    }
  }

  // ---- branches: conv+relu x3, f32 output [p*2+n] ----
  launch_gemm<576, 64, 0, EPI_RELU>(stream, XU,   NB, IbW[0], IbB[0], 64,
                                    T64a, 64, 0, NB, 0, 0, NF, NCF);
  launch_gemm<64, 64, 0, EPI_RELU>(stream, T64a, NB, IbW[1], IbB[1], 64,
                                    T64b, 64, 0, NB, 0, 0, NF, NCF);
  launch_gemm<64, 16, 0, EPI_BRF32>(stream, T64b, NB, IbW[2], IbB[2], 2,
                                    (bf16*)nullptr, 2, 0, NB, 0, 0, BRx, NCF);
  launch_gemm<576, 64, 0, EPI_RELU>(stream, RU,   NB, RbW[0], RbB[0], 64,
                                    T64a, 64, 0, NB, 0, 0, NF, NCF);
  launch_gemm<64, 64, 0, EPI_RELU>(stream, T64a, NB, RbW[1], RbB[1], 64,
                                    T64b, 64, 0, NB, 0, 0, NF, NCF);
  launch_gemm<64, 16, 0, EPI_BRF32>(stream, T64b, NB, RbW[2], RbB[2], 2,
                                    (bf16*)nullptr, 2, 0, NB, 0, 0, BRr, NCF);

  // ---- final: conv_last(q) + branch, written channel-major to d_out ----
  float* outq = (float*)d_out;
  launch_gemm<64, 16, 0, EPI_FINAL>(stream, qbuf[0], NB, LastW, LastB, 2,
                                    (bf16*)nullptr, 0, 0, NB, 2, 0, outq, BRx);
  launch_gemm<64, 16, 0, EPI_FINAL>(stream, rbuf[0], NB, LastW, LastB, 2,
                                    (bf16*)nullptr, 0, 0, NB, 2, 0, outq + 2 * M_PIX, BRr);
}
